// MultiHeadCausalAttention_22462678958385
// MI455X (gfx1250) — compile-verified, matrix operand order checked by reading, not run on hardware
//
#include <hip/hip_runtime.h>
#include <hip/hip_bf16.h>

// MI455X / gfx1250 multi-head causal attention.
// - All matmuls on v_wmma_f32_16x16x32_bf16 (wave32 WMMA), fp32 accumulate.
// - GEMM A-tiles staged into LDS by the Tensor Data Mover (TENSOR_LOAD_TO_LDS,
//   TENSORcnt-synchronized); B-tiles staged transposed by vector loads so
//   B-fragments are contiguous per lane.
// - Flash attention: online softmax with wave32 shfl_xor row reductions,
//   P relaid C-layout -> A-layout through per-wave LDS (s_wait_dscnt).
// Workspace: ~48 MB.

typedef __attribute__((ext_vector_type(16))) __bf16 bf16x16;
typedef __attribute__((ext_vector_type(8)))  __bf16 bf16x8;
typedef __attribute__((ext_vector_type(8)))  float  f32x8;
typedef __attribute__((ext_vector_type(4)))  unsigned int u32x4;
typedef __attribute__((ext_vector_type(8)))  int    i32x8;
typedef __attribute__((ext_vector_type(4)))  int    i32x4;

#define WMMA_BF16(A, B, C) \
  __builtin_amdgcn_wmma_f32_16x16x32_bf16(false, (A), false, (B), (short)0, (C), false, false)

static __device__ __forceinline__ bf16x16 ld2(const __bf16* p0, const __bf16* p1) {
  bf16x16 r;
  *(bf16x8*)&r       = *(const bf16x8*)p0;
  *((bf16x8*)&r + 1) = *(const bf16x8*)p1;
  return r;
}

// --------------------------------------------------------------- TDM load ---
// 2D tile DMA: global (row-major, stride_d0 elements/row) -> LDS (packed).
// D# packing per CDNA5 ISA 8.3/8.4: group0 = {count=1, lds_addr, global_addr,
// type=2}; group1 = {data_size=2B, tensor dims, tile dims, dim0 stride}.
static __device__ __forceinline__ void tdm_load_2d_bf16(
    unsigned lds_off, const void* gaddr,
    unsigned tensor_d0, unsigned tensor_d1,
    unsigned tile_d0, unsigned tile_d1, unsigned stride_d0) {
  const unsigned long long ga = (unsigned long long)gaddr;
  u32x4 g0;
  g0[0] = 1u;                                            // count=1, user mode
  g0[1] = lds_off;                                       // lds_addr
  g0[2] = (unsigned)(ga & 0xffffffffu);                  // global_addr[31:0]
  g0[3] = (unsigned)((ga >> 32) & 0x01ffffffu)           // global_addr[56:32]
          | 0x80000000u;                                 // type = 2 ("image")
  i32x8 g1;
  g1[0] = (int)(1u << 16);                               // data_size=1 -> 2B
  g1[1] = (int)((tensor_d0 & 0xffffu) << 16);            // tensor_dim0 lo
  g1[2] = (int)((tensor_d0 >> 16) | ((tensor_d1 & 0xffffu) << 16));
  g1[3] = (int)((tensor_d1 >> 16) | (tile_d0 << 16));    // dim1 hi | tile_dim0
  g1[4] = (int)(tile_d1 & 0xffffu);                      // tile_dim1, tile_dim2=0
  g1[5] = (int)stride_d0;                                // tensor_dim0_stride lo
  g1[6] = 0;
  g1[7] = 0;
  const i32x4 z = {0, 0, 0, 0};
#if __clang_major__ >= 23
  const i32x8 z8 = {0, 0, 0, 0, 0, 0, 0, 0};
  __builtin_amdgcn_tensor_load_to_lds(g0, g1, z, z, z8, 0);
#else
  __builtin_amdgcn_tensor_load_to_lds(g0, g1, z, z, 0);
#endif
}

// ---------------------------------------------------------------- convert ---
__global__ __launch_bounds__(256) void cvt_f32_bf16_kernel(
    const float* __restrict__ src, __bf16* __restrict__ dst, int n8) {
  int i = blockIdx.x * blockDim.x + threadIdx.x;
  if (i < n8) {
    const float4 v0 = ((const float4*)src)[2 * i];
    const float4 v1 = ((const float4*)src)[2 * i + 1];
    bf16x8 o;
    o[0] = (__bf16)v0.x; o[1] = (__bf16)v0.y; o[2] = (__bf16)v0.z; o[3] = (__bf16)v0.w;
    o[4] = (__bf16)v1.x; o[5] = (__bf16)v1.y; o[6] = (__bf16)v1.z; o[7] = (__bf16)v1.w;
    ((bf16x8*)dst)[i] = o;
  }
}

// ------------------------------------------------------------------- GEMM ---
// C[M,N] = A[M,K](bf16) @ B[K,N](bf16) + bias. Block tile 64x256, BK=32.
// 8 waves: wave (wm,wn) computes 32x64 = 2x4 WMMA 16x16 tiles (8 WMMA/k-iter).
// A tile: TDM DMA to LDS @0; B tile: vector loads, stored transposed @4096.
enum { MODE_F32 = 0, MODE_BHSD = 1, MODE_BHDS = 2 };

template <int MODE>
__global__ __launch_bounds__(256) void gemm_bf16_kernel(
    const __bf16* __restrict__ A, const __bf16* __restrict__ Bm,
    const float* __restrict__ bias, void* __restrict__ outp,
    int M, int N, int K) {
  constexpr int BM = 64, BN = 256, BK = 32;
  extern __shared__ __align__(16) char smem_raw[];
  __bf16(*sA)[BK]      = (__bf16(*)[BK])smem_raw;                       // 4 KB @ 0
  __bf16(*sBT)[BK + 8] = (__bf16(*)[BK + 8])(smem_raw + BM * BK * 2);   // 20 KB @ 4096

  const int tid = threadIdx.x;
  const int n0 = blockIdx.x * BN;
  const int m0 = blockIdx.y * BM;

  const int lane = tid & 31, wave = tid >> 5;
  const int wm = wave >> 2, wn = wave & 3;        // 2 x 4 wave grid
  const int l16 = lane & 15, lh = lane >> 4;
  const int r0 = lh * 8, r1 = 16 + lh * 8;

  const int brow = tid >> 3, bseg = (tid & 7) * 4;  // B: 32 rows x 256 cols

  f32x8 acc[2][4] = {};

  for (int k0 = 0; k0 < K; k0 += BK) {
    // A tile via Tensor Data Mover: 64 rows x 32 cols, row stride K elements.
    if (wave == 0)
      tdm_load_2d_bf16(/*lds_off=*/0u, A + (size_t)m0 * K + k0,
                       (unsigned)K, (unsigned)M, BK, BM, (unsigned)K);
    if (k0 + BK < K)  // global_prefetch_b8 for next B k-tile
      __builtin_prefetch(Bm + (size_t)(k0 + BK + brow) * N + n0 + bseg * 8, 0, 0);
    // Stage B tile transposed (N-major) so B fragments are contiguous per lane.
#pragma unroll
    for (int c = 0; c < 4; ++c) {
      const int bcol = (bseg + c) * 8;
      bf16x8 v = *(const bf16x8*)(Bm + (size_t)(k0 + brow) * N + n0 + bcol);
#pragma unroll
      for (int j = 0; j < 8; ++j) sBT[bcol + j][brow] = v[j];
    }
    if (wave == 0) __builtin_amdgcn_s_wait_tensorcnt((short)0);
    __syncthreads();

    bf16x16 af[2], bfr[4];
#pragma unroll
    for (int t = 0; t < 2; ++t) {
      const int row = wm * 32 + t * 16 + l16;
      af[t] = ld2(&sA[row][r0], &sA[row][r1]);
    }
#pragma unroll
    for (int t = 0; t < 4; ++t) {
      const int col = wn * 64 + t * 16 + l16;
      bfr[t] = ld2(&sBT[col][r0], &sBT[col][r1]);
    }
#pragma unroll
    for (int tm = 0; tm < 2; ++tm)
#pragma unroll
      for (int tn = 0; tn < 4; ++tn)
        acc[tm][tn] = WMMA_BF16(af[tm], bfr[tn], acc[tm][tn]);
    __syncthreads();
  }

  // Epilogue. C layout: VGPR r holds M = r + 8*(lane>=16); lane%16 = N col.
#pragma unroll
  for (int tn = 0; tn < 4; ++tn) {
    const int gn = n0 + wn * 64 + tn * 16 + l16;
    const float bb = bias[gn];
#pragma unroll
    for (int tm = 0; tm < 2; ++tm) {
#pragma unroll
      for (int r = 0; r < 8; ++r) {
        const int gm = m0 + wm * 32 + tm * 16 + r + 8 * lh;
        const float val = acc[tm][tn][r] + bb;
        if constexpr (MODE == MODE_F32) {
          ((float*)outp)[(size_t)gm * N + gn] = val;
        } else {
          const int b = gm >> 11, s = gm & 2047;   // S = 2048
          const int h = gn >> 6,  d = gn & 63;     // Dh = 64
          __bf16* o = (__bf16*)outp;
          if constexpr (MODE == MODE_BHSD)
            o[((size_t)(b * 16 + h) * 2048 + s) * 64 + d] = (__bf16)val;
          else  // MODE_BHDS: V pre-transposed for attention B-fragments
            o[((size_t)(b * 16 + h) * 64 + d) * 2048 + s] = (__bf16)val;
        }
      }
    }
  }
}

// -------------------------------------------------------- flash attention ---
// 1 block = 4 waves = one 64-row Q band of one (b,h). Wave w: 16 q rows.
__global__ __launch_bounds__(128) void attn_causal_kernel(
    const __bf16* __restrict__ Qb, const __bf16* __restrict__ Kb,
    const __bf16* __restrict__ VTb, __bf16* __restrict__ Ctx) {
  constexpr int S = 2048, H = 16, Dh = 64, D = 1024;
  __shared__ alignas(16) __bf16 sP[4][16][40];  // per-wave 16x32 P tile (+pad)

  const int tid = threadIdx.x;
  const int lane = tid & 31, wave = tid >> 5;
  const int l16 = lane & 15, lh = lane >> 4;
  const int r0 = lh * 8, r1 = 16 + lh * 8;

  const int qt = blockIdx.x & 31;
  const int h  = (blockIdx.x >> 5) & 15;
  const int b  = blockIdx.x >> 9;
  const int qbase = qt * 64 + wave * 16;

  const __bf16* Qh = Qb  + (size_t)(b * H + h) * S * Dh;
  const __bf16* Kh = Kb  + (size_t)(b * H + h) * S * Dh;
  const __bf16* Vh = VTb + (size_t)(b * H + h) * Dh * S;

  // Q fragment (A layout, 16x64 = two 16x32 frags), loaded once.
  const int qrow = qbase + l16;
  const bf16x16 aq0 = ld2(Qh + (size_t)qrow * Dh + r0,      Qh + (size_t)qrow * Dh + r1);
  const bf16x16 aq1 = ld2(Qh + (size_t)qrow * Dh + 32 + r0, Qh + (size_t)qrow * Dh + 32 + r1);

  float mrow[8], lrow[8];
  f32x8 o0 = {}, o1 = {}, o2 = {}, o3 = {};
#pragma unroll
  for (int r = 0; r < 8; ++r) { mrow[r] = -3.0e38f; lrow[r] = 0.0f; }

  const int nch = (qbase + 47) >> 5;  // 32-row K chunks up to the diagonal
  for (int kc = 0; kc < nch; ++kc) {
    const int kb = kc * 32;
    const bool masked = (kb + 31) > qbase;

    // K^T B-fragments: lane = score column (k row s-index), d contiguous.
    const int ks0 = kb + l16, ks1 = kb + 16 + l16;
    const bf16x16 bk00 = ld2(Kh + (size_t)ks0 * Dh + r0,      Kh + (size_t)ks0 * Dh + r1);
    const bf16x16 bk01 = ld2(Kh + (size_t)ks0 * Dh + 32 + r0, Kh + (size_t)ks0 * Dh + 32 + r1);
    const bf16x16 bk10 = ld2(Kh + (size_t)ks1 * Dh + r0,      Kh + (size_t)ks1 * Dh + r1);
    const bf16x16 bk11 = ld2(Kh + (size_t)ks1 * Dh + 32 + r0, Kh + (size_t)ks1 * Dh + 32 + r1);

    f32x8 s0 = {}, s1 = {};
    s0 = WMMA_BF16(aq0, bk00, s0); s0 = WMMA_BF16(aq1, bk01, s0);
    s1 = WMMA_BF16(aq0, bk10, s1); s1 = WMMA_BF16(aq1, bk11, s1);

    float alpha[8];
#pragma unroll
    for (int r = 0; r < 8; ++r) {
      const int row = qbase + r + 8 * lh;
      float x0 = s0[r] * 0.125f;  // Dh^-0.5
      float x1 = s1[r] * 0.125f;
      if (masked) {
        if (kb + l16 > row)      x0 = -3.0e38f;
        if (kb + 16 + l16 > row) x1 = -3.0e38f;
      }
      float mx = fmaxf(x0, x1);
#pragma unroll
      for (int sft = 1; sft < 16; sft <<= 1)
        mx = fmaxf(mx, __shfl_xor(mx, sft, 32));
      const float mnew = fmaxf(mrow[r], mx);
      const float a = __expf(mrow[r] - mnew);
      const float e0 = __expf(x0 - mnew);
      const float e1 = __expf(x1 - mnew);
      float rs = e0 + e1;
#pragma unroll
      for (int sft = 1; sft < 16; sft <<= 1)
        rs += __shfl_xor(rs, sft, 32);
      mrow[r] = mnew;
      lrow[r] = lrow[r] * a + rs;
      alpha[r] = a;
      // P into LDS (C layout positions), bf16.
      sP[wave][r + 8 * lh][l16]      = (__bf16)e0;
      sP[wave][r + 8 * lh][16 + l16] = (__bf16)e1;
    }
#pragma unroll
    for (int r = 0; r < 8; ++r) {
      o0[r] *= alpha[r]; o1[r] *= alpha[r];
      o2[r] *= alpha[r]; o3[r] *= alpha[r];
    }

    // Wave-private LDS relayout: only DS ordering needed (CDNA5 split cnt).
    asm volatile("s_wait_dscnt 0" ::: "memory");

    const bf16x16 ap = ld2(&sP[wave][l16][r0], &sP[wave][l16][r1]);

    // V B-fragments from [b][h][d][s]: lane = d column, s contiguous.
    {
      const __bf16* vp = Vh + (size_t)(0 * 16 + l16) * S + kb;
      o0 = WMMA_BF16(ap, ld2(vp + r0, vp + r1), o0);
    }
    {
      const __bf16* vp = Vh + (size_t)(1 * 16 + l16) * S + kb;
      o1 = WMMA_BF16(ap, ld2(vp + r0, vp + r1), o1);
    }
    {
      const __bf16* vp = Vh + (size_t)(2 * 16 + l16) * S + kb;
      o2 = WMMA_BF16(ap, ld2(vp + r0, vp + r1), o2);
    }
    {
      const __bf16* vp = Vh + (size_t)(3 * 16 + l16) * S + kb;
      o3 = WMMA_BF16(ap, ld2(vp + r0, vp + r1), o3);
    }
  }

#pragma unroll
  for (int r = 0; r < 8; ++r) {
    const float inv = 1.0f / lrow[r];
    const int row = qbase + r + 8 * lh;
    __bf16* cp = Ctx + ((size_t)b * S + row) * D + h * Dh;
    cp[0 * 16 + l16] = (__bf16)(o0[r] * inv);
    cp[1 * 16 + l16] = (__bf16)(o1[r] * inv);
    cp[2 * 16 + l16] = (__bf16)(o2[r] * inv);
    cp[3 * 16 + l16] = (__bf16)(o3[r] * inv);
  }
}

// ----------------------------------------------------------------- launch ---
extern "C" void kernel_launch(void* const* d_in, const int* in_sizes, int n_in,
                              void* d_out, int out_size, void* d_ws, size_t ws_size,
                              hipStream_t stream) {
  (void)in_sizes; (void)n_in; (void)out_size; (void)ws_size;
  constexpr int B = 2, S = 2048, D = 1024, M = B * S;

  const float* x  = (const float*)d_in[0];
  const float* Wq = (const float*)d_in[1];
  const float* bq = (const float*)d_in[2];
  const float* Wk = (const float*)d_in[3];
  const float* bk = (const float*)d_in[4];
  const float* Wv = (const float*)d_in[5];
  const float* bv = (const float*)d_in[6];
  const float* Wo = (const float*)d_in[7];
  const float* bo = (const float*)d_in[8];
  float* out = (float*)d_out;

  char* ws = (char*)d_ws;
  __bf16* Xb  = (__bf16*)ws; ws += (size_t)M * D * 2;  //  8 MB
  __bf16* Wqb = (__bf16*)ws; ws += (size_t)D * D * 2;  //  2 MB
  __bf16* Wkb = (__bf16*)ws; ws += (size_t)D * D * 2;
  __bf16* Wvb = (__bf16*)ws; ws += (size_t)D * D * 2;
  __bf16* Wob = (__bf16*)ws; ws += (size_t)D * D * 2;
  __bf16* Qb  = (__bf16*)ws; ws += (size_t)M * D * 2;  // [b][h][s][d]
  __bf16* Kb  = (__bf16*)ws; ws += (size_t)M * D * 2;  // [b][h][s][d]
  __bf16* VTb = (__bf16*)ws; ws += (size_t)M * D * 2;  // [b][h][d][s]
  __bf16* Ctx = (__bf16*)ws; ws += (size_t)M * D * 2;  // [b*s][h*d]

  // fp32 -> bf16 down-conversion (8 elems/thread).
  {
    const int n8x = M * D / 8;
    cvt_f32_bf16_kernel<<<(n8x + 255) / 256, 256, 0, stream>>>(x, Xb, n8x);
    const int n8w = D * D / 8;
    cvt_f32_bf16_kernel<<<(n8w + 255) / 256, 256, 0, stream>>>(Wq, Wqb, n8w);
    cvt_f32_bf16_kernel<<<(n8w + 255) / 256, 256, 0, stream>>>(Wk, Wkb, n8w);
    cvt_f32_bf16_kernel<<<(n8w + 255) / 256, 256, 0, stream>>>(Wv, Wvb, n8w);
    cvt_f32_bf16_kernel<<<(n8w + 255) / 256, 256, 0, stream>>>(Wo, Wob, n8w);
  }

  // Dynamic LDS: A tile (4 KB @ 0, TDM target) + transposed B tile (20 KB).
  constexpr size_t GEMM_LDS = 64 * 32 * 2 + 256 * 40 * 2;
  const dim3 gg(D / 256, M / 64);  // (4, 64) blocks of 256 threads
  gemm_bf16_kernel<MODE_BHSD><<<gg, 256, GEMM_LDS, stream>>>(Xb, Wqb, bq, Qb,  M, D, D);
  gemm_bf16_kernel<MODE_BHSD><<<gg, 256, GEMM_LDS, stream>>>(Xb, Wkb, bk, Kb,  M, D, D);
  gemm_bf16_kernel<MODE_BHDS><<<gg, 256, GEMM_LDS, stream>>>(Xb, Wvb, bv, VTb, M, D, D);

  attn_causal_kernel<<<B * 16 * (S / 64), 128, 0, stream>>>(Qb, Kb, VTb, Ctx);

  gemm_bf16_kernel<MODE_F32><<<gg, 256, GEMM_LDS, stream>>>(Ctx, Wob, bo, out, M, D, D);
}
